// CRF_16758962389301
// MI455X (gfx1250) — matrix operand order repaired and checked by reading, not compiled, not run on hardware
//
#include <hip/hip_runtime.h>

#define B_DIM 128
#define T_DIM 512
#define L_DIM 256
#define NWAVE 8            // waves per block
#define BLOCK (NWAVE * 32) // 256 threads
#define ROWPAD 264         // padded LDS row stride (halves) -> bank-conflict-free A loads

#define LN2F   0.6931471805599453f
#define LOG2EF 1.4426950408889634f

typedef __attribute__((ext_vector_type(16))) _Float16 v16h;
typedef __attribute__((ext_vector_type(8)))  _Float16 v8h;
typedef __attribute__((ext_vector_type(8)))  float    v8f;
typedef __attribute__((ext_vector_type(4)))  unsigned int u32x4;

// ---------------------------------------------------------------------------
// Kernel 1: path_score[b] = sum_t y_pre[b,t,label_t] + sum_t trans[l_t,l_{t+1}]
// Bandwidth-bound (reads the 67MB one-hot y_true once). Writes into d_out.
// ---------------------------------------------------------------------------
__global__ __launch_bounds__(BLOCK) void crf_path_score(
    const float* __restrict__ y_true, const float* __restrict__ y_pre,
    const float* __restrict__ trans, float* __restrict__ out)
{
  const int b   = blockIdx.x;
  const int tid = threadIdx.x;
  __shared__ int   lbl[T_DIM];
  __shared__ float red[BLOCK];

  const size_t base = (size_t)b * T_DIM * L_DIM;
  for (int t = 0; t < T_DIM; ++t) {
    float v = y_true[base + (size_t)t * L_DIM + tid]; // coalesced over tid
    if (v > 0.5f) lbl[t] = tid;                        // exactly one writer
  }
  __syncthreads();

  float acc = 0.f;
  for (int t = tid; t < T_DIM; t += BLOCK)
    acc += y_pre[base + (size_t)t * L_DIM + lbl[t]];
  for (int t = tid; t < T_DIM - 1; t += BLOCK)
    acc += trans[lbl[t] * L_DIM + lbl[t + 1]];

  red[tid] = acc;
  __syncthreads();
  for (int s = BLOCK / 2; s > 0; s >>= 1) {
    if (tid < s) red[tid] += red[tid + s];
    __syncthreads();
  }
  if (tid == 0) out[b] = red[0];
}

// ---------------------------------------------------------------------------
// Kernel 2: forward algorithm in the exp domain using v_wmma_f32_16x16x32_f16.
// One block = 16 batch rows (M=16). 8 waves, each owns 32 output columns
// (two 16x16 C tiles). B = exp(trans) is held entirely in VGPRs (step-
// invariant); A = exp(state - rowmax) is rebuilt in LDS every step.
// Transcendentals use raw v_log_f32 / v_exp_f32: the log argument is
// provably in [0.6, ~420] (each expA row contains an exact 1.0 and
// expT >= exp(min trans)), so no denormal guards are needed.
// ---------------------------------------------------------------------------
__global__ __launch_bounds__(BLOCK) void crf_forward(
    const float* __restrict__ y_pre, const float* __restrict__ trans,
    float* __restrict__ out)
{
  __shared__ __align__(16) _Float16 sExpA[16][ROWPAD]; // exp(state - off), f16
  __shared__ __align__(16) float    sPart[NWAVE][16];  // per-wave row partials
  __shared__ __align__(16) float    sMaxq[16];         // per-row step max

  const int tid   = threadIdx.x;
  const int lane  = tid & 31;
  const int w     = tid >> 5;
  const int half  = lane >> 4;      // 0: low K/M half, 1: high half
  const int l15   = lane & 15;
  const int b0    = blockIdx.x * 16;
  const int n0    = w * 32;
  const int nc0   = n0 + l15;       // column of C tile 0 owned by this lane
  const int nc1   = n0 + 16 + l15;  // column of C tile 1
  const int r     = l15;            // A-matrix row this lane loads
  const int rbase = half * 8;       // first C row this lane owns

  // ---- preload B operand: B[K][N] = exp(trans[K*256+N]) (step-invariant) --
  // 16-bit B striping: lanes 0-15 hold K = c*32..+15, lanes 16-31 K = +16,
  // two K-values packed per dword.
  v16h bt0[8], bt1[8];
  #pragma unroll
  for (int c = 0; c < 8; ++c) {
    const int kb = c * 32 + half * 16;
    v16h x0, x1;
    #pragma unroll
    for (int i = 0; i < 16; ++i) {
      x0[i] = (_Float16)__builtin_amdgcn_exp2f(
                  trans[(size_t)(kb + i) * L_DIM + nc0] * LOG2EF);
      x1[i] = (_Float16)__builtin_amdgcn_exp2f(
                  trans[(size_t)(kb + i) * L_DIM + nc1] * LOG2EF);
    }
    bt0[c] = x0; bt1[c] = x1;
  }

  // per-lane emission base offsets (8 rows x 2 column tiles)
  size_t em0[8], em1[8];
  #pragma unroll
  for (int j = 0; j < 8; ++j) {
    const size_t rb = (size_t)(b0 + rbase + j) * T_DIM * (size_t)L_DIM;
    em0[j] = rb + nc0;
    em1[j] = rb + nc1;
  }

  float e0[8], e1[8]; // prefetched emissions for current step
  #pragma unroll
  for (int j = 0; j < 8; ++j) { e0[j] = y_pre[em0[j]]; e1[j] = y_pre[em1[j]]; }

  float runOff = 0.f;     // running row offset; valid in threads 0..15
  float ex0[8], ex1[8];   // exp(state - off) from the latest step (registers)

  for (int t = 0; t < T_DIM; ++t) {
    float q0[8], q1[8];   // q = new_state - prevOff (log s + emit)
    if (t == 0) {
      #pragma unroll
      for (int j = 0; j < 8; ++j) { q0[j] = e0[j]; q1[j] = e1[j]; }
    } else {
      __syncthreads();    // expA(t-1) visible to all waves
      v8f a0 = {0.f, 0.f, 0.f, 0.f, 0.f, 0.f, 0.f, 0.f};
      v8f a1 = a0;
      #pragma unroll
      for (int c = 0; c < 8; ++c) {
        const int ka = c * 32 + half * 16; // A 16x32 f16: contiguous K = 16*half + 0..15
        v8h lo = __builtin_bit_cast(v8h, *(const u32x4*)&sExpA[r][ka]);
        v8h hi = __builtin_bit_cast(v8h, *(const u32x4*)&sExpA[r][ka + 8]);
        v16h av = __builtin_shufflevector(lo, hi,
            0,1,2,3,4,5,6,7,8,9,10,11,12,13,14,15);
        a0 = __builtin_amdgcn_wmma_f32_16x16x32_f16(false, av, false, bt0[c],
                                                    (short)0, a0, false, false);
        a1 = __builtin_amdgcn_wmma_f32_16x16x32_f16(false, av, false, bt1[c],
                                                    (short)0, a1, false, false);
      }
      // q = ln(s) + emit  ==  ln2 * log2(s) + emit  (raw v_log_f32 + v_fma)
      #pragma unroll
      for (int j = 0; j < 8; ++j) {
        q0[j] = fmaf(LN2F, __builtin_amdgcn_logf(a0[j]), e0[j]);
        q1[j] = fmaf(LN2F, __builtin_amdgcn_logf(a1[j]), e1[j]);
      }
    }

    // per-row max over this wave's 32 columns (xor stays within 16-lane half)
    float m8[8];
    #pragma unroll
    for (int j = 0; j < 8; ++j) m8[j] = fmaxf(q0[j], q1[j]);
    #pragma unroll
    for (int mk = 8; mk >= 1; mk >>= 1) {
      #pragma unroll
      for (int j = 0; j < 8; ++j)
        m8[j] = fmaxf(m8[j], __shfl_xor(m8[j], mk, 32));
    }
    if (l15 == 0) {
      #pragma unroll
      for (int j = 0; j < 8; ++j) sPart[w][rbase + j] = m8[j];
    }
    __syncthreads();
    if (tid < 16) {
      float mq = sPart[0][tid];
      #pragma unroll
      for (int ww = 1; ww < NWAVE; ++ww) mq = fmaxf(mq, sPart[ww][tid]);
      runOff += mq;
      sMaxq[tid] = mq;
    }
    __syncthreads();

    // rebuild A = exp(q - rowmax) in LDS (f16), keep f32 copy in registers
    // exp(x) == exp2(x * log2e); argument <= 0, underflow-to-zero is correct.
    #pragma unroll
    for (int j = 0; j < 8; ++j) {
      const float mq = sMaxq[rbase + j];
      ex0[j] = __builtin_amdgcn_exp2f((q0[j] - mq) * LOG2EF);
      ex1[j] = __builtin_amdgcn_exp2f((q1[j] - mq) * LOG2EF);
      sExpA[rbase + j][nc0] = (_Float16)ex0[j];
      sExpA[rbase + j][nc1] = (_Float16)ex1[j];
    }

    // prefetch next step's emissions while others are still in the exp phase
    if (t + 1 < T_DIM) {
      const size_t off = (size_t)(t + 1) * L_DIM;
      #pragma unroll
      for (int j = 0; j < 8; ++j) {
        e0[j] = y_pre[em0[j] + off];
        e1[j] = y_pre[em1[j] + off];
      }
    }
  }

  // final log-sum-exp over columns: log_norm[row] = runOff + ln(sum exp(.))
  float s8[8];
  #pragma unroll
  for (int j = 0; j < 8; ++j) s8[j] = ex0[j] + ex1[j];
  #pragma unroll
  for (int mk = 8; mk >= 1; mk >>= 1) {
    #pragma unroll
    for (int j = 0; j < 8; ++j)
      s8[j] += __shfl_xor(s8[j], mk, 32);
  }
  __syncthreads();
  if (l15 == 0) {
    #pragma unroll
    for (int j = 0; j < 8; ++j) sPart[w][rbase + j] = s8[j];
  }
  __syncthreads();
  if (tid < 16) {
    float tot = sPart[0][tid];
    #pragma unroll
    for (int ww = 1; ww < NWAVE; ++ww) tot += sPart[ww][tid];
    const float ps = out[b0 + tid];          // path score from kernel 1
    out[b0 + tid] = fmaf(LN2F, __builtin_amdgcn_logf(tot), runOff) - ps;
  }
}

// ---------------------------------------------------------------------------
extern "C" void kernel_launch(void* const* d_in, const int* in_sizes, int n_in,
                              void* d_out, int out_size, void* d_ws, size_t ws_size,
                              hipStream_t stream) {
  (void)in_sizes; (void)n_in; (void)out_size; (void)d_ws; (void)ws_size;
  const float* y_true = (const float*)d_in[0];
  const float* y_pre  = (const float*)d_in[1];
  const float* trans  = (const float*)d_in[2];
  float* outp = (float*)d_out;

  crf_path_score<<<B_DIM, BLOCK, 0, stream>>>(y_true, y_pre, trans, outp);
  crf_forward<<<B_DIM / 16, BLOCK, 0, stream>>>(y_pre, trans, outp);
}
